// MaskedGNNModel_30313879176001
// MI455X (gfx1250) — matrix operand order repaired and checked by reading, not compiled, not run on hardware
//
#include <hip/hip_runtime.h>
#include <hip/hip_bf16.h>

// ---------------------------------------------------------------------------
// GCN (3-layer) for MI455X / gfx1250.
//   h1 = relu(Ahat @ (x  W1) + b1)
//   h2 = relu(Ahat @ (h1 W2) + b2)
//   out = log_softmax(h2 W3 + b3)
// Ahat = D^-1/2 (A + I) D^-1/2.
// Dense transforms use v_wmma_f32_16x16x32_bf16 (bf16 in, fp32 accumulate).
// Sparse aggregation = one wave per edge, atomicAdd into L2-resident [N,64].
// ---------------------------------------------------------------------------

typedef __attribute__((ext_vector_type(16))) __bf16 v16bf;
typedef __attribute__((ext_vector_type(8)))  float  v8f;

#define IN_DIM 256
#define HID    64
#define OUTC   40
#define OUTP   48   // 40 padded to 3 * 16

// ---------------------------------------------------------------- utilities
__global__ void fill_kernel(float* __restrict__ p, float v, long n) {
  long i = (long)blockIdx.x * blockDim.x + threadIdx.x;
  if (i < n) p[i] = v;
}

__global__ void deg_count_kernel(const int* __restrict__ dst,
                                 float* __restrict__ deg, int E) {
  int e = blockIdx.x * blockDim.x + threadIdx.x;
  if (e < E) atomicAdd(&deg[dst[e]], 1.0f);
}

__global__ void dinv_kernel(const float* __restrict__ deg,
                            float* __restrict__ dinv, int n) {
  int i = blockIdx.x * blockDim.x + threadIdx.x;
  if (i < n) dinv[i] = rsqrtf(deg[i]);   // deg >= 1 (self loop)
}

// Pre-swizzle a [K, Ncols] fp32 weight matrix into CDNA5 WMMA B-fragment
// order (bf16). Fragment (kt, nt) = 32 lanes x 16 halves; lane n = N column,
// lanes 0-15 hold K = kt*32 + 0..7,16..23, lanes 16-31 hold K = kt*32 + 8..15,24..31.
__global__ void swizzleW_kernel(const float* __restrict__ W,
                                __bf16* __restrict__ out,
                                int KT, int NT, int Ncols) {
  int t = blockIdx.x * blockDim.x + threadIdx.x;
  int total = KT * NT * 512;
  if (t >= total) return;
  int frag   = t >> 9;
  int within = t & 511;
  int lane = within >> 4;
  int i    = within & 15;
  int kt = frag / NT;
  int nt = frag - kt * NT;
  int k = kt * 32 + ((lane >= 16) ? 8 : 0) + (i & 7) + ((i >= 8) ? 16 : 0);
  int n = nt * 16 + (lane & 15);
  float v = (n < Ncols) ? W[k * Ncols + n] : 0.0f;
  out[t] = (__bf16)v;
}

// ---------------------------------------------------------------- WMMA GEMM
// C[M, NT*16] = A[M, KDIM] (fp32, converted to bf16) @ Bswz (pre-swizzled).
// One wave per 16-row tile; 8 waves / block. M must be a multiple of 16
// (100000 = 6250 * 16).
template <int KDIM, int NT>
__global__ __launch_bounds__(256)
void gemm_wmma_kernel(const float* __restrict__ A,
                      const __bf16* __restrict__ Bswz,
                      float* __restrict__ C, int M) {
  const int lane = threadIdx.x & 31;
  const int wave = threadIdx.x >> 5;
  const int tile = blockIdx.x * 8 + wave;
  const int row0 = tile * 16;
  if (row0 >= M) return;                      // wave-uniform exit: EXEC stays full

  // A fragment addressing (16-bit A 16x32 layout):
  //   lanes 0-15 : M = lane,    a[0..7] = K0..7,  a[8..15] = K16..23
  //   lanes 16-31: M = lane-16, a[0..7] = K8..15, a[8..15] = K24..31
  const int m     = row0 + (lane & 15);
  const int kbase = (lane >= 16) ? 8 : 0;
  const float* __restrict__ arow = A + (long)m * KDIM;
  const v16bf* __restrict__ bw = (const v16bf*)Bswz;

  v8f acc[NT] = {};

#pragma unroll
  for (int kt = 0; kt < KDIM / 32; ++kt) {
    const int k0 = kt * 32 + kbase;
    float4 f0 = *(const float4*)(arow + k0);
    float4 f1 = *(const float4*)(arow + k0 + 4);
    float4 f2 = *(const float4*)(arow + k0 + 16);
    float4 f3 = *(const float4*)(arow + k0 + 20);
    v16bf a;
    a[0]  = (__bf16)f0.x; a[1]  = (__bf16)f0.y; a[2]  = (__bf16)f0.z; a[3]  = (__bf16)f0.w;
    a[4]  = (__bf16)f1.x; a[5]  = (__bf16)f1.y; a[6]  = (__bf16)f1.z; a[7]  = (__bf16)f1.w;
    a[8]  = (__bf16)f2.x; a[9]  = (__bf16)f2.y; a[10] = (__bf16)f2.z; a[11] = (__bf16)f2.w;
    a[12] = (__bf16)f3.x; a[13] = (__bf16)f3.y; a[14] = (__bf16)f3.z; a[15] = (__bf16)f3.w;
#pragma unroll
    for (int nt = 0; nt < NT; ++nt) {
      v16bf b = bw[(kt * NT + nt) * 32 + lane];
      acc[nt] = __builtin_amdgcn_wmma_f32_16x16x32_bf16(
          /*neg_a=*/false, a, /*neg_b=*/false, b,
          /*c_mod=*/(short)0, acc[nt], /*reuse_a=*/false, /*reuse_b=*/false);
    }
  }

  // C/D layout: VGPR r, lane l -> row = row0 + r + (l>=16 ? 8 : 0), col = nt*16 + (l&15)
  const int rowhi = (lane >> 4) * 8;
#pragma unroll
  for (int nt = 0; nt < NT; ++nt) {
    const int col = nt * 16 + (lane & 15);
#pragma unroll
    for (int r = 0; r < 8; ++r) {
      const int row = row0 + rowhi + r;
      C[(long)row * (NT * 16) + col] = acc[nt][r];
    }
  }
}

// ---------------------------------------------------------- edge aggregation
// One wave per edge; lane handles 2 of the 64 features. Source rows and
// destination rows are L2-resident (25.6 MB each << 192 MB L2).
__global__ __launch_bounds__(256)
void edge_agg_kernel(const float* __restrict__ h,
                     const int* __restrict__ src, const int* __restrict__ dst,
                     const float* __restrict__ dinv,
                     float* __restrict__ out, int E) {
  int e = blockIdx.x * 8 + (threadIdx.x >> 5);
  if (e >= E) return;
  int s = src[e];
  int d = dst[e];
  float w = dinv[s] * dinv[d];
  int j = (threadIdx.x & 31) * 2;
  float2 v = *(const float2*)(h + (long)s * HID + j);
  atomicAdd(out + (long)d * HID + j,     w * v.x);
  atomicAdd(out + (long)d * HID + j + 1, w * v.y);
}

// agg[i,j] = relu?( agg[i,j] + dinv[i]^2 * h[i,j] + bias[j] )   (in place)
__global__ void self_bias_relu_kernel(float* __restrict__ agg,
                                      const float* __restrict__ h,
                                      const float* __restrict__ dinv,
                                      const float* __restrict__ bias,
                                      long n64, int relu) {
  long idx = (long)blockIdx.x * blockDim.x + threadIdx.x;
  if (idx >= n64) return;
  int i = (int)(idx >> 6);
  int j = (int)(idx & 63);
  float di = dinv[i];
  float v = agg[idx] + di * di * h[idx] + bias[j];
  if (relu) v = fmaxf(v, 0.0f);
  agg[idx] = v;
}

// log_softmax over 40 real columns of the padded [N,48] logits, + b3.
__global__ void logsoftmax_kernel(const float* __restrict__ logits,
                                  const float* __restrict__ b3,
                                  float* __restrict__ out, int n) {
  int i = blockIdx.x * blockDim.x + threadIdx.x;
  if (i >= n) return;
  const float* row = logits + (long)i * OUTP;
  float mx = -3.0e38f;
  for (int j = 0; j < OUTC; ++j) mx = fmaxf(mx, row[j] + b3[j]);
  float s = 0.0f;
  for (int j = 0; j < OUTC; ++j) s += __expf(row[j] + b3[j] - mx);
  float ls = __logf(s);
  float* orow = out + (long)i * OUTC;
  for (int j = 0; j < OUTC; ++j) orow[j] = row[j] + b3[j] - mx - ls;
}

// ---------------------------------------------------------------- launcher
extern "C" void kernel_launch(void* const* d_in, const int* in_sizes, int n_in,
                              void* d_out, int out_size, void* d_ws, size_t ws_size,
                              hipStream_t stream) {
  (void)n_in; (void)out_size; (void)ws_size;
  const float* x  = (const float*)d_in[0];
  const int*   ei = (const int*)d_in[1];      // [2, E]
  const float* W1 = (const float*)d_in[2];
  const float* b1 = (const float*)d_in[3];
  const float* W2 = (const float*)d_in[4];
  const float* b2 = (const float*)d_in[5];
  const float* W3 = (const float*)d_in[6];
  const float* b3 = (const float*)d_in[7];
  float* out = (float*)d_out;

  const int N = in_sizes[0] / IN_DIM;
  const int E = in_sizes[1] / 2;
  const int* src = ei;
  const int* dst = ei + E;

  // Workspace carve-up (256B aligned regions)
  char* ws = (char*)d_ws;
  size_t off = 0;
  auto carve = [&](size_t bytes) -> void* {
    void* p = ws + off;
    off = (off + bytes + 255) & ~(size_t)255;
    return p;
  };
  float*  deg  = (float*)carve((size_t)N * 4);
  float*  dinv = (float*)carve((size_t)N * 4);
  float*  bufA = (float*)carve((size_t)N * HID * 4);   // gemm outputs / logits
  float*  bufB = (float*)carve((size_t)N * HID * 4);   // aggregation targets
  __bf16* W1s  = (__bf16*)carve((size_t)IN_DIM * HID * 2);
  __bf16* W2s  = (__bf16*)carve((size_t)HID * HID * 2);
  __bf16* W3s  = (__bf16*)carve((size_t)HID * OUTP * 2);

  const long n64 = (long)N * HID;
  const int TPB = 256;
  const int gN    = (N + TPB - 1) / TPB;
  const int gE    = (E + TPB - 1) / TPB;
  const int gN64  = (int)((n64 + TPB - 1) / TPB);
  const int gGemm = ((N / 16) + 7) / 8;                // 16-row tile per wave, 8 waves/block
  const int gEdge = (E + 7) / 8;                       // 1 wave per edge

  // Weight pre-swizzle into WMMA B-fragment layout
  swizzleW_kernel<<<(IN_DIM / 32 * 4 * 512 + TPB - 1) / TPB, TPB, 0, stream>>>(W1, W1s, IN_DIM / 32, 4, HID);
  swizzleW_kernel<<<(HID / 32 * 4 * 512 + TPB - 1) / TPB, TPB, 0, stream>>>(W2, W2s, HID / 32, 4, HID);
  swizzleW_kernel<<<(HID / 32 * 3 * 512 + TPB - 1) / TPB, TPB, 0, stream>>>(W3, W3s, HID / 32, 3, OUTC);

  // Degree (with self loop) and D^-1/2
  fill_kernel<<<gN, TPB, 0, stream>>>(deg, 1.0f, N);
  deg_count_kernel<<<gE, TPB, 0, stream>>>(dst, deg, E);
  dinv_kernel<<<gN, TPB, 0, stream>>>(deg, dinv, N);

  // Layer 1: transform -> aggregate -> bias+relu
  gemm_wmma_kernel<IN_DIM, 4><<<gGemm, TPB, 0, stream>>>(x, W1s, bufA, N);
  fill_kernel<<<gN64, TPB, 0, stream>>>(bufB, 0.0f, n64);
  edge_agg_kernel<<<gEdge, TPB, 0, stream>>>(bufA, src, dst, dinv, bufB, E);
  self_bias_relu_kernel<<<gN64, TPB, 0, stream>>>(bufB, bufA, dinv, b1, n64, 1);

  // Layer 2
  gemm_wmma_kernel<HID, 4><<<gGemm, TPB, 0, stream>>>(bufB, W2s, bufA, N);
  fill_kernel<<<gN64, TPB, 0, stream>>>(bufB, 0.0f, n64);
  edge_agg_kernel<<<gEdge, TPB, 0, stream>>>(bufA, src, dst, dinv, bufB, E);
  self_bias_relu_kernel<<<gN64, TPB, 0, stream>>>(bufB, bufA, dinv, b2, n64, 1);

  // Layer 3: transform (padded to 48 cols) + log_softmax
  gemm_wmma_kernel<HID, 3><<<gGemm, TPB, 0, stream>>>(bufB, W3s, bufA, N);
  logsoftmax_kernel<<<gN, TPB, 0, stream>>>(bufA, b3, out, N);
}
